// Model_39676907887182
// MI455X (gfx1250) — matrix operand order repaired and checked by reading, not compiled, not run on hardware
//
#include <hip/hip_runtime.h>

typedef __attribute__((ext_vector_type(16))) _Float16 v16h;
typedef __attribute__((ext_vector_type(8)))  _Float16 v8h;
typedef __attribute__((ext_vector_type(8)))  float    v8f;

#define S_LEN  2048
#define D_DIM  128
#define MB     256        // query rows per workgroup (8 waves x 32 rows)
#define NW     8
#define QS_STR 136        // padded LDS strides (f16 elements)
#define KS_STR 136
#define VS_STR 40
#define SS_STR 40

// LDS arena layout (bytes)
#define KB_OFF   0                           // 2 x 32 x KS_STR f16  = 17408
#define VB_OFF   17408                       // 2 x 128 x VS_STR f16 = 20480
#define SS_OFF   (17408 + 20480)             // 8 x 2 x 16 x SS_STR  = 20480
#define ARENA_SZ (MB * QS_STR * 2)           // Q staging dominates: 69632

__device__ __forceinline__ v16h ld_frag(const _Float16* p0, const _Float16* p1) {
  v8h lo = *(const v8h*)p0;
  v8h hi = *(const v8h*)p1;
  return __builtin_shufflevector(lo, hi, 0,1,2,3,4,5,6,7,8,9,10,11,12,13,14,15);
}

__global__ __launch_bounds__(256) void fused_qkmaskv_kernel(
    const float* __restrict__ q, const float* __restrict__ k,
    const float* __restrict__ v, const float* __restrict__ mask,
    float* __restrict__ out)
{
  __shared__ alignas(16) char arena[ARENA_SZ];
  _Float16* Qs  = (_Float16*)arena;               // staging phase only
  _Float16* Kb  = (_Float16*)(arena + KB_OFF);    // [buf][32][KS_STR]
  _Float16* Vb  = (_Float16*)(arena + VB_OFF);    // [buf][128][VS_STR] (transposed)
  _Float16* Ssb = (_Float16*)(arena + SS_OFF);    // [wave][mt][16][SS_STR]

  const int tid  = threadIdx.x;
  const int lane = tid & 31;
  const int wave = tid >> 5;
  const int ln   = lane & 15;
  const int hl   = lane >> 4;

  const int mBlocks = S_LEN / MB;                 // 8
  const int bh = blockIdx.x / mBlocks;
  const int mb = blockIdx.x % mBlocks;

  const float* qBH = q    + (size_t)bh * S_LEN * D_DIM;
  const float* kBH = k    + (size_t)bh * S_LEN * D_DIM;
  const float* vBH = v    + (size_t)bh * S_LEN * D_DIM;
  const float* mBH = mask + (size_t)bh * S_LEN * S_LEN;
  float*       oBH = out  + (size_t)bh * S_LEN * D_DIM;

  const int m0 = mb * MB;

  // ---- stage Q block: 256x128 fp32 -> f16 LDS (32 float4 / thread) ----
  #pragma unroll
  for (int i = 0; i < 32; ++i) {
    int f   = tid + i * 256;
    int row = f >> 5;
    int c   = (f & 31) << 2;
    float4 x = *(const float4*)(qBH + (size_t)(m0 + row) * D_DIM + c);
    _Float16 t4[4] = {(_Float16)x.x, (_Float16)x.y, (_Float16)x.z, (_Float16)x.w};
    *(unsigned long long*)&Qs[row * QS_STR + c] = *(unsigned long long*)t4;
  }
  __syncthreads();

  // Two 16-row A-fragment tiles per wave, 4 K-chunks each
  v16h qf[2][4];
  #pragma unroll
  for (int mt = 0; mt < 2; ++mt)
    #pragma unroll
    for (int kc = 0; kc < 4; ++kc) {
      const _Float16* r = &Qs[(wave * 32 + mt * 16 + ln) * QS_STR + kc * 32];
      qf[mt][kc] = ld_frag(r + hl * 8, r + 16 + hl * 8);
    }
  __syncthreads();   // everyone's Q in registers; arena now owned by K/V/S

  v8f o[2][8] = {};

  // ---- software-pipelined K/V staging (double-buffered in LDS) ----
  float4 kr[4], vr[4];
  int srow[4], scol[4];
  #pragma unroll
  for (int i = 0; i < 4; ++i) {
    int f = tid + i * 256;
    srow[i] = f >> 5;            // 0..31 key row
    scol[i] = (f & 31) << 2;     // feature col (float4 granularity)
  }

  // prologue: global loads for tile 0
  #pragma unroll
  for (int i = 0; i < 4; ++i) {
    kr[i] = *(const float4*)(kBH + (size_t)srow[i] * D_DIM + scol[i]);
    vr[i] = *(const float4*)(vBH + (size_t)srow[i] * D_DIM + scol[i]);
  }

  for (int it = 0; it < S_LEN / 32; ++it) {
    const int t0  = it * 32;
    const int buf = it & 1;
    _Float16* Kt = Kb + buf * 32 * KS_STR;
    _Float16* Vt = Vb + buf * D_DIM * VS_STR;

    // 1) store staged tile (regs -> LDS), K row-major, V transposed
    #pragma unroll
    for (int i = 0; i < 4; ++i) {
      _Float16 t4[4] = {(_Float16)kr[i].x, (_Float16)kr[i].y,
                        (_Float16)kr[i].z, (_Float16)kr[i].w};
      *(unsigned long long*)&Kt[srow[i] * KS_STR + scol[i]] = *(unsigned long long*)t4;
      Vt[(scol[i] + 0) * VS_STR + srow[i]] = (_Float16)vr[i].x;
      Vt[(scol[i] + 1) * VS_STR + srow[i]] = (_Float16)vr[i].y;
      Vt[(scol[i] + 2) * VS_STR + srow[i]] = (_Float16)vr[i].z;
      Vt[(scol[i] + 3) * VS_STR + srow[i]] = (_Float16)vr[i].w;
    }
    // 2) issue global loads for next tile (hidden behind compute below)
    if (it + 1 < S_LEN / 32) {
      const int tn = t0 + 32;
      #pragma unroll
      for (int i = 0; i < 4; ++i) {
        kr[i] = *(const float4*)(kBH + (size_t)(tn + srow[i]) * D_DIM + scol[i]);
        vr[i] = *(const float4*)(vBH + (size_t)(tn + srow[i]) * D_DIM + scol[i]);
      }
    }
    // 3) single barrier: tile `it` visible, prior reads of this buf retired
    __syncthreads();

    // 4) S = Q @ K^T : each K B-fragment feeds BOTH M-tiles
    #pragma unroll
    for (int j = 0; j < 2; ++j) {
      v8f s[2] = {};
      #pragma unroll
      for (int kc = 0; kc < 4; ++kc) {
        const _Float16* r = &Kt[(16 * j + ln) * KS_STR + kc * 32 + hl * 8];
        v16h bf = ld_frag(r, r + 16);
        s[0] = __builtin_amdgcn_wmma_f32_16x16x32_f16(false, qf[0][kc], false, bf,
                                                      (short)0, s[0], false, false);
        s[1] = __builtin_amdgcn_wmma_f32_16x16x32_f16(false, qf[1][kc], false, bf,
                                                      (short)0, s[1], false, false);
      }
      // dropout mask (NT stream) + f16 stash, per M-tile
      #pragma unroll
      for (int mt = 0; mt < 2; ++mt) {
        _Float16* sw = Ssb + (wave * 2 + mt) * 16 * SS_STR;
        const size_t mrowBase =
            (size_t)(m0 + wave * 32 + mt * 16 + 8 * hl) * S_LEN
            + (size_t)(t0 + 16 * j + ln);
        #pragma unroll
        for (int rr = 0; rr < 8; ++rr) {
          float mv = __builtin_nontemporal_load(mBH + mrowBase + (size_t)rr * S_LEN);
          sw[(rr + 8 * hl) * SS_STR + 16 * j + ln] = (_Float16)(s[mt][rr] * mv);
        }
      }
    }

    // 5) O += S' @ V : each V B-fragment feeds BOTH M-tiles
    v16h af[2];
    #pragma unroll
    for (int mt = 0; mt < 2; ++mt) {
      const _Float16* sr = Ssb + (wave * 2 + mt) * 16 * SS_STR + ln * SS_STR;
      af[mt] = ld_frag(sr + hl * 8, sr + 16 + hl * 8);
    }
    #pragma unroll
    for (int nc = 0; nc < 8; ++nc) {
      const _Float16* r = &Vt[(nc * 16 + ln) * VS_STR + hl * 8];
      v16h bf = ld_frag(r, r + 16);
      o[0][nc] = __builtin_amdgcn_wmma_f32_16x16x32_f16(false, af[0], false, bf,
                                                        (short)0, o[0][nc], false, false);
      o[1][nc] = __builtin_amdgcn_wmma_f32_16x16x32_f16(false, af[1], false, bf,
                                                        (short)0, o[1][nc], false, false);
    }
  }

  // ---- write 32x128 output rows per wave (C layout -> row-major, NT) ----
  #pragma unroll
  for (int mt = 0; mt < 2; ++mt)
    #pragma unroll
    for (int nc = 0; nc < 8; ++nc) {
      size_t base = (size_t)(m0 + wave * 32 + mt * 16 + 8 * hl) * D_DIM
                    + nc * 16 + ln;
      #pragma unroll
      for (int rr = 0; rr < 8; ++rr)
        __builtin_nontemporal_store(o[mt][nc][rr], oBH + base + (size_t)rr * D_DIM);
    }
}

extern "C" void kernel_launch(void* const* d_in, const int* in_sizes, int n_in,
                              void* d_out, int out_size, void* d_ws, size_t ws_size,
                              hipStream_t stream) {
  const float* q    = (const float*)d_in[0];
  const float* k    = (const float*)d_in[1];
  const float* v    = (const float*)d_in[2];
  const float* mask = (const float*)d_in[3];
  float* out        = (float*)d_out;

  const int BH = in_sizes[0] / (S_LEN * D_DIM);   // 32
  dim3 grid(BH * (S_LEN / MB));                   // 256 workgroups
  dim3 block(256);                                // 8 wave32 waves
  fused_qkmaskv_kernel<<<grid, block, 0, stream>>>(q, k, v, mask, out);
}
